// EmbedMatcher_32195074851391
// MI455X (gfx1250) — matrix operand order repaired and checked by reading, not compiled, not run on hardware
//
#include <hip/hip_runtime.h>
#include <hip/hip_bf16.h>
#include <stdint.h>

// ---------------------------------------------------------------------------
// Types / WMMA helpers (CDNA5 gfx1250, wave32)
// ---------------------------------------------------------------------------
typedef __bf16 bf16;
typedef __attribute__((ext_vector_type(16))) __bf16 v16bf;
typedef __attribute__((ext_vector_type(8)))  float  v8f;
typedef unsigned int u32;
typedef __attribute__((ext_vector_type(4))) u32 u32x4;
typedef __attribute__((ext_vector_type(4))) int i32x4;
typedef __attribute__((ext_vector_type(8))) int i32x8;

struct Frag32 { uint4 lo, hi; };

__device__ __forceinline__ v8f wmma_bf16(v16bf a, v16bf b, v8f c) {
  // D = A(16x32 bf16) * B(32x16 bf16) + C(16x16 f32)
  return __builtin_amdgcn_wmma_f32_16x16x32_bf16(false, a, false, b, (short)0, c,
                                                 false, false);
}

// A fragment (16x32 bf16) from row-major storage. Per ISA 7.12.2:
// lanes 0-15: row M=lane, K={0..7,16..23}; lanes 16-31: row M=lane-16, K={8..15,24..31}.
// Caller passes p0 = &base[row*stride + half*8]; chunks are K+0..7 and K+16..23.
__device__ __forceinline__ v16bf load_afrag(const bf16* p0) {
  Frag32 t;
  t.lo = *(const uint4*)(p0);
  t.hi = *(const uint4*)(p0 + 16);
  return __builtin_bit_cast(v16bf, t);
}
// B fragment (32x16 bf16) from [N][K] row-major storage: lane holds column N=lane%16,
// K = {half*8..+7, 16+half*8..+7} (same as A).  p0 = &base[col*stride + half*8].
__device__ __forceinline__ v16bf load_bfrag(const bf16* p0) {
  Frag32 t;
  t.lo = *(const uint4*)(p0);
  t.hi = *(const uint4*)(p0 + 16);
  return __builtin_bit_cast(v16bf, t);
}

// ---------------------------------------------------------------------------
// Tensor Data Mover (CDNA5): one instruction DMAs a 2D tile global->LDS with
// hardware row padding. Guarded: falls back to cooperative copies if absent.
// ---------------------------------------------------------------------------
#if defined(__has_builtin)
#if __has_builtin(__builtin_amdgcn_tensor_load_to_lds)
#define HAVE_TDM 1
#endif
#endif

#define GBM 128
#define GBN 128
#define GBK 32
#define GLDT 40  // padded LDS row stride (halves): 80B rows keep 16B alignment

#if defined(HAVE_TDM)
__device__ __forceinline__ u32 lds_addr32(const void* p) {
  // flat shared address: low 32 bits are the LDS byte offset (ISA 10.2 truncation)
  return (u32)(size_t)p;
}

// Load a (tile_rows x 32) bf16 tile; LDS rows padded 64B->80B via TDM pad fields.
// rows_avail < tile_rows => hardware zero-fill of trailing rows (tensor OOB reads).
__device__ __forceinline__ void tdm_load_tile_b16(u32 lds_byte_addr,
                                                  const bf16* gsrc,
                                                  int tile_rows,
                                                  int rows_avail,
                                                  int row_stride_elems)
{
  const unsigned long long ga = (unsigned long long)(size_t)gsrc;
  u32x4 g0;
  g0[0] = 1u;                                              // count=1, user mode
  g0[1] = lds_byte_addr;                                   // lds_addr
  g0[2] = (u32)(ga & 0xffffffffull);                       // global_addr[31:0]
  g0[3] = (u32)((ga >> 32) & 0x01ffffffull) | (2u << 30);  // addr[56:32] | type=2
  const u32 td0 = 32u;                                     // tensor_dim0 (elems)
  const u32 td1 = (u32)(rows_avail > 0 ? rows_avail : 0);  // tensor_dim1
  const unsigned long long s0 = (unsigned long long)(long long)row_stride_elems;
  i32x8 g1;
  // data_size=2B | pad_enable | pad_interval=3 (16 DW = 64B row) | pad_amount=3 (4 DW = 16B)
  g1[0] = (int)((1u << 16) | (1u << 20) | (3u << 22) | (3u << 25));
  g1[1] = (int)((td0 & 0xffffu) << 16);                           // tensor_dim0[15:0]
  g1[2] = (int)(((td0 >> 16) & 0xffffu) | ((td1 & 0xffffu) << 16));
  g1[3] = (int)(((td1 >> 16) & 0xffffu) | (td0 << 16));           // tile_dim0 = 32
  g1[4] = (int)(tile_rows & 0xffff);                              // tile_dim1; tile_dim2=0
  g1[5] = (int)(s0 & 0xffffffffull);                              // dim0 stride (elems)
  g1[6] = (int)((s0 >> 32) & 0xffffull);                          // stride[47:32]; d1stride=0
  g1[7] = 0;
  i32x4 g2; g2[0] = 0; g2[1] = 0; g2[2] = 0; g2[3] = 0;
  i32x4 g3 = g2;
#if defined(__clang_major__) && (__clang_major__ >= 23)
  i32x8 g4;
  g4[0]=0; g4[1]=0; g4[2]=0; g4[3]=0; g4[4]=0; g4[5]=0; g4[6]=0; g4[7]=0;
  __builtin_amdgcn_tensor_load_to_lds(g0, g1, g2, g3, g4, 0);
#else
  __builtin_amdgcn_tensor_load_to_lds(g0, g1, g2, g3, 0);
#endif
}
#endif  // HAVE_TDM

// ---------------------------------------------------------------------------
// Generic tiled bf16 WMMA GEMM:  C[M,N] = act(A[M,K] @ B[N,K]^T + bias + add)
// 256 threads = 8 waves, block tile 128x128, BK=32, wave tile 32x64.
// Double-buffered LDS; tiles moved by TDM (tensor_load_to_lds) when available.
// ---------------------------------------------------------------------------
__global__ __launch_bounds__(256)
void wmma_gemm_bf16_kernel(const bf16* __restrict__ A, int lda,
                           const bf16* __restrict__ B, int ldb,
                           float* __restrict__ C, int ldc,
                           bf16* __restrict__ Cbf, int ldcbf,
                           const float* __restrict__ bias,
                           const float* __restrict__ addsrc, int ldadd,
                           int M, int N, int K, int act)
{
  __shared__ __align__(16) bf16 As[2][GBM * GLDT];
  __shared__ __align__(16) bf16 Bs[2][GBN * GLDT];
  const int tid  = threadIdx.x;
  const int lane = tid & 31, wid = tid >> 5;
  const int lm   = lane & 15, half = lane >> 4;
  const int tilesN = (N + GBN - 1) / GBN;
  const int m0 = (blockIdx.x / tilesN) * GBM;
  const int n0 = (blockIdx.x % tilesN) * GBN;
  const int wr = wid & 3;   // 32-row group
  const int wc = wid >> 2;  // 64-col group
  const int ktiles = K / GBK;

  v8f acc[2][4];
#pragma unroll
  for (int i = 0; i < 2; ++i)
#pragma unroll
    for (int j = 0; j < 4; ++j)
#pragma unroll
      for (int v = 0; v < 8; ++v) acc[i][j][v] = 0.0f;

#if defined(HAVE_TDM)
  if (wid == 0) {
    tdm_load_tile_b16(lds_addr32(&As[0][0]), A + (size_t)m0 * lda, GBM, M - m0, lda);
    tdm_load_tile_b16(lds_addr32(&Bs[0][0]), B + (size_t)n0 * ldb, GBN, N - n0, ldb);
    __builtin_amdgcn_s_wait_tensorcnt(0);
  }
  __syncthreads();
#else
  const int r  = tid >> 1;           // 0..127 tile row loaded by this thread
  const int kc = (tid & 1) * 16;     // half-row of 32 halves
  int ga = m0 + r; if (ga >= M) ga = M - 1;  // clamp: dup rows discarded on store
  int gb = n0 + r; if (gb >= N) gb = N - 1;
  const bf16* pa = A + (size_t)ga * lda + kc;
  const bf16* pb = B + (size_t)gb * ldb + kc;
  uint4 ra0 = *(const uint4*)(pa), ra1 = *(const uint4*)(pa + 8);
  uint4 rb0 = *(const uint4*)(pb), rb1 = *(const uint4*)(pb + 8);
  *(uint4*)&As[0][r * GLDT + kc]     = ra0;
  *(uint4*)&As[0][r * GLDT + kc + 8] = ra1;
  *(uint4*)&Bs[0][r * GLDT + kc]     = rb0;
  *(uint4*)&Bs[0][r * GLDT + kc + 8] = rb1;
  __syncthreads();
#endif

  for (int kt = 0; kt < ktiles; ++kt) {
    const int cur = kt & 1, nxt = cur ^ 1;
#if defined(HAVE_TDM)
    if (wid == 0 && kt + 1 < ktiles) {
      const int ko = (kt + 1) * GBK;
      tdm_load_tile_b16(lds_addr32(&As[nxt][0]), A + (size_t)m0 * lda + ko, GBM, M - m0, lda);
      tdm_load_tile_b16(lds_addr32(&Bs[nxt][0]), B + (size_t)n0 * ldb + ko, GBN, N - n0, ldb);
    }
#else
    if (kt + 1 < ktiles) {
      const int ko = (kt + 1) * GBK;
      ra0 = *(const uint4*)(pa + ko); ra1 = *(const uint4*)(pa + ko + 8);
      rb0 = *(const uint4*)(pb + ko); rb1 = *(const uint4*)(pb + ko + 8);
    }
#endif
    v16bf af[2], bfv[4];
#pragma unroll
    for (int i = 0; i < 2; ++i)
      af[i] = load_afrag(&As[cur][(wr * 32 + i * 16 + lm) * GLDT + half * 8]);
#pragma unroll
    for (int j = 0; j < 4; ++j)
      bfv[j] = load_bfrag(&Bs[cur][(wc * 64 + j * 16 + lm) * GLDT + half * 8]);
#pragma unroll
    for (int i = 0; i < 2; ++i)
#pragma unroll
      for (int j = 0; j < 4; ++j)
        acc[i][j] = wmma_bf16(af[i], bfv[j], acc[i][j]);
#if defined(HAVE_TDM)
    if (wid == 0) __builtin_amdgcn_s_wait_tensorcnt(0);
#else
    if (kt + 1 < ktiles) {
      *(uint4*)&As[nxt][r * GLDT + kc]     = ra0;
      *(uint4*)&As[nxt][r * GLDT + kc + 8] = ra1;
      *(uint4*)&Bs[nxt][r * GLDT + kc]     = rb0;
      *(uint4*)&Bs[nxt][r * GLDT + kc + 8] = rb1;
    }
#endif
    __syncthreads();
  }

  // Epilogue. C/D layout: element v of lane: M = v + 8*(lane/16), N = lane%16.
#pragma unroll
  for (int i = 0; i < 2; ++i) {
#pragma unroll
    for (int j = 0; j < 4; ++j) {
      const int gn = n0 + wc * 64 + j * 16 + lm;
      if (gn >= N) continue;
      const float bv = bias ? bias[gn] : 0.0f;
#pragma unroll
      for (int v = 0; v < 8; ++v) {
        const int gm = m0 + wr * 32 + i * 16 + half * 8 + v;
        if (gm >= M) continue;
        float x = acc[i][j][v] + bv;
        if (addsrc) x += addsrc[(size_t)gm * ldadd + gn];
        if (act == 1) x = x > 0.0f ? x : 0.0f;
        if (C)   C[(size_t)gm * ldc + gn] = x;
        if (Cbf) Cbf[(size_t)gm * ldcbf + gn] = (bf16)x;
      }
    }
  }
}

// ---------------------------------------------------------------------------
// Neighbor GCN encoder: one workgroup per node n.
//   cat[k] = [emb[rel_k], emb[ent_k]] (512)  -> proj = leaky(cat @ gcn_W^T + wb + b)
//   masked sum over k / max(deg,1); gate = sigmoid(mean(gate_w[rel])/temp) (1 if deg==0)
//   out = tanh(emb[self] + gate*agg)  -> writes vec row (f32 + bf16 mirror)
// Two passes of 32 neighbors staged as bf16 in LDS; deterministic LDS partial sums.
// ---------------------------------------------------------------------------
#define PAD_ID 400000
#define CATLD 520  // 512 + 8 pad halves (1040B rows: 16B aligned)

__global__ __launch_bounds__(256)
void neighbor_encoder_kernel(const float* __restrict__ emb,
                             const bf16* __restrict__ Wbf,   // [256][512] bf16
                             const float* __restrict__ wb,
                             const float* __restrict__ bb,
                             const float* __restrict__ gate_w,
                             const float* __restrict__ gate_temp,
                             const int* __restrict__ conn,   // [n][64][2]
                             const int* __restrict__ deg,
                             const int* __restrict__ selfids, int selfStride, int selfOff,
                             float* __restrict__ outF, bf16* __restrict__ outBf,
                             int rowOffset, int colOffset)
{
  __shared__ __align__(16) bf16 cat[32 * CATLD];
  __shared__ float aggP[4 * 256];
  __shared__ float gvals[64];
  __shared__ int relL[64], entL[64], padL[64];

  const int n = blockIdx.x;
  const int tid = threadIdx.x;
  const int lane = tid & 31, wid = tid >> 5;
  const int lm = lane & 15, half = lane >> 4;
  const int wr = wid & 1;   // 16-row group within the 32-row pass
  const int wc = wid >> 1;  // 64-col group (4 * 64 = 256)

  if (tid < 64) {
    const int rel = conn[(n * 64 + tid) * 2 + 0];
    const int ent = conn[(n * 64 + tid) * 2 + 1];
    relL[tid] = rel; entL[tid] = ent;
    padL[tid] = (rel == PAD_ID) ? 1 : 0;
    gvals[tid] = gate_w[(rel == PAD_ID) ? 0 : rel];
  }
  for (int i = tid; i < 1024; i += 256) aggP[i] = 0.0f;

  for (int pass = 0; pass < 2; ++pass) {
    __syncthreads();
    // gather 64 emb rows (32 rel + 32 ent) -> cat bf16
    for (int it = 0; it < 16; ++it) {
      const int idx = tid + it * 256;      // 0..4095 float4 slots
      const int row = idx >> 6;            // 0..63
      const int c4  = idx & 63;
      const int k   = row & 31;
      const int kk  = pass * 32 + k;
      const int id  = (row < 32) ? relL[kk] : entL[kk];
      const float4 v = *(const float4*)&emb[(size_t)id * 256 + c4 * 4];
      bf16* d = &cat[k * CATLD + ((row < 32) ? 0 : 256) + c4 * 4];
      d[0] = (bf16)v.x; d[1] = (bf16)v.y; d[2] = (bf16)v.z; d[3] = (bf16)v.w;
    }
    __syncthreads();
    // 32x512 @ W^T -> 32x256, wave tile 16x64
    v8f acc[4];
#pragma unroll
    for (int t = 0; t < 4; ++t)
#pragma unroll
      for (int v = 0; v < 8; ++v) acc[t][v] = 0.0f;
    for (int kt = 0; kt < 512; kt += 32) {
      const v16bf a = load_afrag(&cat[(wr * 16 + lm) * CATLD + kt + half * 8]);
#pragma unroll
      for (int t = 0; t < 4; ++t) {
        const int col = wc * 64 + t * 16 + lm;
        const v16bf b = load_bfrag(&Wbf[(size_t)col * 512 + kt + half * 8]);
        acc[t] = wmma_bf16(a, b, acc[t]);
      }
    }
    // masked leaky-relu + row-sum into unique (slot,col) partial -> deterministic
#pragma unroll
    for (int t = 0; t < 4; ++t) {
      const int col = wc * 64 + t * 16 + lm;
      const float bv = wb[col] + bb[col];
      float s = 0.0f;
#pragma unroll
      for (int v = 0; v < 8; ++v) {
        const int m = wr * 16 + half * 8 + v;   // 0..31 within pass
        float x = acc[t][v] + bv;
        x = x > 0.0f ? x : 0.01f * x;           // jax leaky_relu slope 0.01
        if (!padL[pass * 32 + m]) s += x;
      }
      aggP[(wr * 2 + half) * 256 + col] += s;   // single writer per (slot,col)
    }
  }
  __syncthreads();
  if (tid < 256) {
    float gs = 0.0f;
#pragma unroll 8
    for (int k = 0; k < 64; ++k) gs += gvals[k];
    const float degf  = (float)deg[n];
    const float denom = degf > 1.0f ? degf : 1.0f;
    float g = 1.0f / (1.0f + expf(-(gs * (1.0f / 64.0f)) / gate_temp[0]));
    if (!(degf > 0.0f)) g = 1.0f;
    const float agg = (aggP[tid] + aggP[256 + tid] + aggP[512 + tid] + aggP[768 + tid]) / denom;
    const int self = selfids[n * selfStride + selfOff];
    const float r = tanhf(emb[(size_t)self * 256 + tid] + g * agg);
    const size_t o = (size_t)(rowOffset + n) * 512 + colOffset + tid;
    outF[o]  = r;
    outBf[o] = (bf16)r;
  }
}

// ---------------------------------------------------------------------------
// Row LayerNorm (512 cols, ddof=1): one wave32 per row.
// ---------------------------------------------------------------------------
__global__ __launch_bounds__(256)
void layernorm_kernel(const float* __restrict__ Z,
                      const float* __restrict__ a, const float* __restrict__ b,
                      float* __restrict__ outF, bf16* __restrict__ outBf, int rows)
{
  const int gw = (int)((blockIdx.x * blockDim.x + threadIdx.x) >> 5);
  const int lane = threadIdx.x & 31;
  if (gw >= rows) return;
  const float* z = Z + (size_t)gw * 512;
  float v[16]; float s = 0.0f;
#pragma unroll
  for (int i = 0; i < 16; ++i) { v[i] = z[lane + i * 32]; s += v[i]; }
#pragma unroll
  for (int o = 16; o > 0; o >>= 1) s += __shfl_xor(s, o, 32);
  const float mu = s * (1.0f / 512.0f);
  float q = 0.0f;
#pragma unroll
  for (int i = 0; i < 16; ++i) { const float d = v[i] - mu; q += d * d; }
#pragma unroll
  for (int o = 16; o > 0; o >>= 1) q += __shfl_xor(q, o, 32);
  const float inv = 1.0f / (sqrtf(q * (1.0f / 511.0f)) + 0.001f);
#pragma unroll
  for (int i = 0; i < 16; ++i) {
    const int c = lane + i * 32;
    const float r = (v[i] - mu) * inv * a[c] + b[c];
    outF[(size_t)gw * 512 + c]  = r;
    outBf[(size_t)gw * 512 + c] = (bf16)r;
  }
}

// ---------------------------------------------------------------------------
// Small helper kernels
// ---------------------------------------------------------------------------
__global__ void sg_mean_kernel(const float* __restrict__ enc, float* __restrict__ sg)
{
  const int c = threadIdx.x;  // blockDim = 512
  float s = 0.0f;
  for (int r = 0; r < 5; ++r) s += enc[(size_t)(2048 + r) * 512 + c];
  sg[c] = s * 0.2f;
}

__global__ void bias_combine_kernel(const float* __restrict__ bih,
                                    const float* __restrict__ bhh,
                                    float* __restrict__ out)
{
  const int n = blockIdx.x * blockDim.x + threadIdx.x;
  if (n < 4096) out[n] = bih[n] + bhh[n];
}

// hh_sg[n] = sum_k s_g[k] * W_hh[n][512+k]  (the loop-invariant h_r[:,512:] term)
__global__ __launch_bounds__(256)
void hh_sg_kernel(const float* __restrict__ Whh, const float* __restrict__ sg,
                  float* __restrict__ out)
{
  __shared__ float sgl[512];
  for (int i = threadIdx.x; i < 512; i += 256) sgl[i] = sg[i];
  __syncthreads();
  const int n = blockIdx.x * 256 + threadIdx.x;
  if (n >= 4096) return;
  const float* w = Whh + (size_t)n * 1024 + 512;
  float s = 0.0f;
  for (int k = 0; k < 512; ++k) s += sgl[k] * w[k];
  out[n] = s;
}

// strided f32 -> bf16 (weight prep; also extracts W_hh[:, :512] packed)
__global__ void cvt_bf16_kernel(const float* __restrict__ in, bf16* __restrict__ out,
                                int rows, int incols, int takecols, int coloff)
{
  const int idx = blockIdx.x * blockDim.x + threadIdx.x;
  if (idx >= rows * takecols) return;
  const int r = idx / takecols, c = idx % takecols;
  out[(size_t)r * takecols + c] = (bf16)in[(size_t)r * incols + coloff + c];
}

// LSTM cell: gates (2048x4096) -> c (2048x1024), h = q_enc + sig(o)*tanh(c) [:512]
__global__ __launch_bounds__(256)
void lstm_cell_kernel(const float* __restrict__ gates, const float* __restrict__ qenc,
                      float* __restrict__ c, float* __restrict__ h, bf16* __restrict__ hbf,
                      int first)
{
  const int idx = blockIdx.x * 256 + threadIdx.x;   // < 2048*1024
  const int m = idx >> 10, j = idx & 1023;
  const float* g = gates + (size_t)m * 4096;
  const float iv = g[j], fv = g[1024 + j], gv = g[2048 + j], ov = g[3072 + j];
  const float si = 1.0f / (1.0f + expf(-iv));
  const float sf = 1.0f / (1.0f + expf(-fv));
  const float so = 1.0f / (1.0f + expf(-ov));
  const float cold = first ? 0.0f : c[idx];
  const float cn = sf * cold + si * tanhf(gv);
  c[idx] = cn;
  if (j < 512) {
    const float hv = qenc[(size_t)m * 512 + j] + so * tanhf(cn);
    h[(size_t)m * 512 + j]  = hv;
    hbf[(size_t)m * 512 + j] = (bf16)hv;
  }
}

// out[m] = dot(h[m], s_g), one wave32 per row
__global__ void final_dot_kernel(const float* __restrict__ h, const float* __restrict__ sg,
                                 float* __restrict__ out)
{
  const int m = blockIdx.x, lane = threadIdx.x;  // blockDim = 32
  float s = 0.0f;
#pragma unroll
  for (int i = 0; i < 16; ++i) {
    const int c = lane + i * 32;
    s += h[(size_t)m * 512 + c] * sg[c];
  }
#pragma unroll
  for (int o = 16; o > 0; o >>= 1) s += __shfl_xor(s, o, 32);
  if (lane == 0) out[m] = s;
}

// ---------------------------------------------------------------------------
// Host launcher
// ---------------------------------------------------------------------------
extern "C" void kernel_launch(void* const* d_in, const int* in_sizes, int n_in,
                              void* d_out, int out_size, void* d_ws, size_t ws_size,
                              hipStream_t stream)
{
  (void)in_sizes; (void)n_in; (void)out_size; (void)ws_size;
  const float* emb       = (const float*)d_in[0];
  const float* gcn_W     = (const float*)d_in[1];
  const float* gcn_wb    = (const float*)d_in[2];
  const float* gcn_b     = (const float*)d_in[3];
  const float* gate_w    = (const float*)d_in[4];
  const float* gate_temp = (const float*)d_in[5];
  const float* p1_W      = (const float*)d_in[6];
  const float* p1_b      = (const float*)d_in[7];
  const float* p2_W      = (const float*)d_in[8];
  const float* p2_b      = (const float*)d_in[9];
  const float* ln_a      = (const float*)d_in[10];
  const float* ln_b      = (const float*)d_in[11];
  const float* W_ih      = (const float*)d_in[12];
  const float* W_hh      = (const float*)d_in[13];
  const float* b_ih      = (const float*)d_in[14];
  const float* b_hh      = (const float*)d_in[15];
  const int* query    = (const int*)d_in[16];
  const int* support  = (const int*)d_in[17];
  const int* q_l1     = (const int*)d_in[18];
  const int* q_deg_l  = (const int*)d_in[19];
  const int* q_r1     = (const int*)d_in[20];
  const int* q_deg_r  = (const int*)d_in[21];
  const int* s_l1     = (const int*)d_in[22];
  const int* s_deg_l  = (const int*)d_in[23];
  const int* s_r1     = (const int*)d_in[24];
  const int* s_deg_r  = (const int*)d_in[25];
  float* out = (float*)d_out;

  char* wsb = (char*)d_ws;
  size_t off = 0;
  auto alloc = [&](size_t bytes) -> void* {
    off = (off + 255) & ~(size_t)255;
    void* p = wsb + off; off += bytes; return p;
  };
  const int NR = 2053;  // 2048 query rows + 5 support rows batched
  float* vecF   = (float*)alloc((size_t)NR * 512 * 4);
  bf16*  vecBf  = (bf16*) alloc((size_t)NR * 512 * 2);
  bf16*  t1Bf   = (bf16*) alloc((size_t)NR * 1024 * 2);
  float* zF     = (float*)alloc((size_t)NR * 512 * 4);   // pre-LN, LN'd in place
  bf16*  encBf  = (bf16*) alloc((size_t)NR * 512 * 2);
  float* sg     = (float*)alloc(512 * 4);
  float* bihhh  = (float*)alloc(4096 * 4);
  float* hhsg   = (float*)alloc(4096 * 4);
  float* gbase  = (float*)alloc((size_t)2048 * 4096 * 4);
  float* gbuf   = (float*)alloc((size_t)2048 * 4096 * 4);
  float* cSt    = (float*)alloc((size_t)2048 * 1024 * 4);
  float* hF     = (float*)alloc((size_t)2048 * 512 * 4);
  bf16*  hBf    = (bf16*) alloc((size_t)2048 * 512 * 2);
  bf16*  gcnWbf = (bf16*) alloc((size_t)256 * 512 * 2);
  bf16*  p1Wbf  = (bf16*) alloc((size_t)1024 * 512 * 2);
  bf16*  p2Wbf  = (bf16*) alloc((size_t)512 * 1024 * 2);
  bf16*  WihBf  = (bf16*) alloc((size_t)4096 * 512 * 2);
  bf16*  WhhLbf = (bf16*) alloc((size_t)4096 * 512 * 2);
  float* encF   = zF;  // LN in place (wave reads row to regs before writing)

  auto cdiv = [](int a, int b) { return (a + b - 1) / b; };

  // 1) weight prep (f32 -> bf16 once per launch)
  cvt_bf16_kernel<<<cdiv(256 * 512, 256), 256, 0, stream>>>(gcn_W, gcnWbf, 256, 512, 512, 0);
  cvt_bf16_kernel<<<cdiv(1024 * 512, 256), 256, 0, stream>>>(p1_W, p1Wbf, 1024, 512, 512, 0);
  cvt_bf16_kernel<<<cdiv(512 * 1024, 256), 256, 0, stream>>>(p2_W, p2Wbf, 512, 1024, 1024, 0);
  cvt_bf16_kernel<<<cdiv(4096 * 512, 256), 256, 0, stream>>>(W_ih, WihBf, 4096, 512, 512, 0);
  cvt_bf16_kernel<<<cdiv(4096 * 512, 256), 256, 0, stream>>>(W_hh, WhhLbf, 4096, 1024, 512, 0);

  // 2) neighbor encoders -> vec rows [q | s], cols [left | right]
  neighbor_encoder_kernel<<<2048, 256, 0, stream>>>(emb, gcnWbf, gcn_wb, gcn_b, gate_w, gate_temp,
                                                    q_l1, q_deg_l, query, 2, 0, vecF, vecBf, 0, 0);
  neighbor_encoder_kernel<<<2048, 256, 0, stream>>>(emb, gcnWbf, gcn_wb, gcn_b, gate_w, gate_temp,
                                                    q_r1, q_deg_r, query, 2, 1, vecF, vecBf, 0, 256);
  neighbor_encoder_kernel<<<5, 256, 0, stream>>>(emb, gcnWbf, gcn_wb, gcn_b, gate_w, gate_temp,
                                                 s_l1, s_deg_l, support, 2, 0, vecF, vecBf, 2048, 0);
  neighbor_encoder_kernel<<<5, 256, 0, stream>>>(emb, gcnWbf, gcn_wb, gcn_b, gate_w, gate_temp,
                                                 s_r1, s_deg_r, support, 2, 1, vecF, vecBf, 2048, 256);

  // 3) support encoder on all 2053 rows: relu(x@p1^T+b1) -> @p2^T+b2+x -> LN
  wmma_gemm_bf16_kernel<<<cdiv(NR, 128) * cdiv(1024, 128), 256, 0, stream>>>(
      vecBf, 512, p1Wbf, 512, nullptr, 0, t1Bf, 1024,
      p1_b, nullptr, 0, NR, 1024, 512, 1);
  wmma_gemm_bf16_kernel<<<cdiv(NR, 128) * cdiv(512, 128), 256, 0, stream>>>(
      t1Bf, 1024, p2Wbf, 1024, zF, 512, nullptr, 0,
      p2_b, vecF, 512, NR, 512, 1024, 0);
  layernorm_kernel<<<cdiv(NR, 8), 256, 0, stream>>>(zF, ln_a, ln_b, encF, encBf, NR);

  // 4) s_g and loop-invariant LSTM biases
  sg_mean_kernel<<<1, 512, 0, stream>>>(encF, sg);
  bias_combine_kernel<<<cdiv(4096, 256), 256, 0, stream>>>(b_ih, b_hh, bihhh);
  hh_sg_kernel<<<16, 256, 0, stream>>>(W_hh, sg, hhsg);

  // 5) gates_base = q_enc @ W_ih^T + (b_ih + b_hh)   (loop-invariant)
  wmma_gemm_bf16_kernel<<<cdiv(2048, 128) * cdiv(4096, 128), 256, 0, stream>>>(
      encBf, 512, WihBf, 512, gbase, 4096, nullptr, 0,
      bihhh, nullptr, 0, 2048, 4096, 512, 0);

  // 6) step 1 (h_r = 0): gates == gates_base
  lstm_cell_kernel<<<(2048 * 1024) / 256, 256, 0, stream>>>(gbase, encF, cSt, hF, hBf, 1);
  // steps 2..4: gates = gates_base + hh_sg + h @ W_hh[:, :512]^T
  for (int s = 1; s < 4; ++s) {
    wmma_gemm_bf16_kernel<<<cdiv(2048, 128) * cdiv(4096, 128), 256, 0, stream>>>(
        hBf, 512, WhhLbf, 512, gbuf, 4096, nullptr, 0,
        hhsg, gbase, 4096, 2048, 4096, 512, 0);
    lstm_cell_kernel<<<(2048 * 1024) / 256, 256, 0, stream>>>(gbuf, encF, cSt, hF, hBf, 0);
  }

  // 7) out[m] = dot(h[m], s_g)
  final_dot_kernel<<<2048, 32, 0, stream>>>(hF, sg, out);
}
